// MultiViewEncoder_8306466750856
// MI455X (gfx1250) — matrix operand order repaired and checked by reading, not compiled, not run on hardware
//
#include <hip/hip_runtime.h>
#include <hip/hip_bf16.h>
#include <cstdint>

typedef __attribute__((ext_vector_type(16))) _Float16 v16h;
typedef __attribute__((ext_vector_type(8)))  _Float16 v8h;
typedef __attribute__((ext_vector_type(8)))  float    v8f;
typedef __attribute__((ext_vector_type(4)))  int      v4i;

#define NIMG 64
#define NGRP 16

#if defined(__AMDGCN__) && __has_builtin(__builtin_amdgcn_global_load_async_to_lds_b128) && __has_builtin(__builtin_amdgcn_s_wait_asynccnt)
#define ASYNC_OK 1
#else
#define ASYNC_OK 0
#endif

__device__ __forceinline__ float selu_f(float x) {
    const float scale = 1.0507009873554805f;
    const float alpha = 1.6732632423543772f;
    return x > 0.f ? scale * x : scale * alpha * (__expf(x) - 1.f);
}
__device__ __forceinline__ int clampi(int x, int lo, int hi) {
    return x < lo ? lo : (x > hi ? hi : x);
}

// ---------------------------------------------------------------------------
// Layout conversion kernels
// ---------------------------------------------------------------------------
// x: NCHW f32 -> NHWC f16
__global__ void cvt_x_nhwc(const float* __restrict__ src, _Float16* __restrict__ dst,
                           int C, int HW, int total) {
    int i = blockIdx.x * 256 + threadIdx.x;
    if (i >= total) return;
    int c = i % C;
    int np = i / C;
    int p = np % HW;
    int n = np / HW;
    dst[i] = (_Float16)src[((size_t)n * C + c) * HW + p];
}
// w: OIHW f32 -> O(HW)I f16  (K ordered as (kh,kw,ci), matching NHWC activations)
__global__ void cvt_w_ohwi(const float* __restrict__ src, _Float16* __restrict__ dst,
                           int Cin, int KHW, int total) {
    int i = blockIdx.x * 256 + threadIdx.x;
    if (i >= total) return;
    int ci = i % Cin;
    int t  = i / Cin;
    int f  = t % KHW;
    int co = t / KHW;
    dst[i] = (_Float16)src[((size_t)co * Cin + ci) * KHW + f];
}

// ---------------------------------------------------------------------------
// Implicit-GEMM conv (NHWC), WMMA f16->f32, geometry fully compile-time.
//   block = 256 thr = 8 waves; tile M=32 x N=128, K-chunk 64.
//   wave (mw,nw): rows [mw*16,+16), cols [nw*32,+32) -> 2 accumulators,
//   4 v_wmma_f32_16x16x32_f16 per chunk per wave; double-buffered LDS.
//   A tile: global_load_async_to_lds_b128 when K%64==0.
//   B tile: async when 1x1/PAD0 & Cin%32==0; vector+select when Cin%32==0;
//           branch-free element path otherwise.
// ---------------------------------------------------------------------------
template <int Cin, int H, int W, int Cout, int KH, int KW, int STRIDE, int PAD>
__global__ __launch_bounds__(256)
void conv_wmma(const _Float16* __restrict__ in, const _Float16* __restrict__ wt,
               const float* __restrict__ bias, const _Float16* __restrict__ resid,
               _Float16* __restrict__ out, int act, int has_resid)
{
    constexpr int K    = Cin * KH * KW;
    constexpr int KHW  = KH * KW;
    constexpr int OH   = (H + 2 * PAD - KH) / STRIDE + 1;
    constexpr int OW   = (W + 2 * PAD - KW) / STRIDE + 1;
    constexpr int OHW  = OH * OW;
    constexpr int NTOT = NIMG * OHW;
    constexpr int KC   = 64;
    constexpr int NCH  = (K + KC - 1) / KC;
    constexpr int AS_K = KC + 8;     // 144B rows: 16B aligned, bank-skewed
    constexpr int BS_K = KC + 8;
    constexpr bool BIGC = (Cin % 32 == 0);
#if ASYNC_OK
    constexpr bool AASYNC = (K % KC == 0);
    constexpr bool BASYNC = BIGC && (KHW == 1) && (PAD == 0);
#else
    constexpr bool AASYNC = false;
    constexpr bool BASYNC = false;
#endif
    constexpr bool ANY_ASYNC = AASYNC || BASYNC;

    __shared__ __align__(16) _Float16 As[2][32 * AS_K];
    __shared__ __align__(16) _Float16 Bs[2][128 * BS_K];

    const int tid  = threadIdx.x;
    const int lane = tid & 31;
    const int wave = tid >> 5;
    const int mw = wave & 1;
    const int nw = wave >> 1;
    const int mbase  = blockIdx.y << 5;
    const int nblock = blockIdx.x << 7;

    // ---- B loader mapping ----
    const int bn  = tid & 127;           // column within tile
    const int bko = (tid >> 7) << 5;     // 0 or 32 within K chunk
    const int gcolB = nblock + bn;
    const bool colOK = gcolB < NTOT;
    const int gcolC = colOK ? gcolB : 0;
    const int imgB = gcolC / OHW;
    const int remB = gcolC - imgB * OHW;
    const int ohB  = remB / OW;
    const int owB  = remB - ohB * OW;
    const int ihb  = ohB * STRIDE - PAD;
    const int iwb  = owB * STRIDE - PAD;
    const _Float16* __restrict__ inB = in + (size_t)imgB * H * W * Cin;
    // 1x1/PAD0: fixed spatial base, contiguous over K
    const _Float16* __restrict__ inSpat = inB + ((size_t)ihb * W + iwb) * Cin;

    // ---- A loader mapping ----
    const int am  = tid >> 3;            // 0..31
    const int ako = (tid & 7) << 3;      // 0..56
    const _Float16* __restrict__ wrow = wt + (size_t)(mbase + am) * K;

    // ---- fragment coords ----
    const int lm = lane & 15;
    const int lk = lane >> 4;

    v8f acc0 = {}, acc1 = {};

    auto stage = [&](int buf, int k0) {
        // ---- A tile ----
        if constexpr (AASYNC) {
#if ASYNC_OK
            __builtin_amdgcn_global_load_async_to_lds_b128(
                (v4i*)(wrow + k0 + ako),
                (v4i*)(&As[buf][am * AS_K + ako]), 0, 0);
#endif
        } else {
            _Float16 av[8];
            #pragma unroll
            for (int u = 0; u < 8; u++) {
                int kg = k0 + ako + u;
                int kc = kg < K - 1 ? kg : K - 1;
                _Float16 v = wrow[kc];
                av[u] = (kg < K && (mbase + am) < Cout) ? v : (_Float16)0.f;
            }
            *(v8h*)(&As[buf][am * AS_K + ako]) = *(v8h*)av;
        }
        // ---- B tile ----
        if constexpr (BASYNC) {
#if ASYNC_OK
            if (colOK) {
                const _Float16* src = inSpat + k0 + bko;
                _Float16* dstl = &Bs[buf][bn * BS_K + bko];
                #pragma unroll
                for (int q = 0; q < 4; q++)
                    __builtin_amdgcn_global_load_async_to_lds_b128(
                        (v4i*)(src + 8 * q), (v4i*)(dstl + 8 * q), 0, 0);
            }
#endif
        } else if constexpr (BIGC) {
            // one contiguous 32-half run within a single (kh,kw) cell; K%64==0
            int kg0 = k0 + bko;
            int f   = kg0 / Cin;
            int ci0 = kg0 - f * Cin;
            int kh = f / KW, kw = f - kh * KW;
            int ih = ihb + kh, iw = iwb + kw;
            bool inb = colOK &&
                       ((PAD == 0) || ((unsigned)ih < (unsigned)H && (unsigned)iw < (unsigned)W));
            int ihc = (PAD == 0) ? ih : clampi(ih, 0, H - 1);
            int iwc = (PAD == 0) ? iw : clampi(iw, 0, W - 1);
            const v8h* src = (const v8h*)(inB + ((size_t)ihc * W + iwc) * Cin + ci0);
            v8h z = {};
            v8h d0 = inb ? src[0] : z;
            v8h d1 = inb ? src[1] : z;
            v8h d2 = inb ? src[2] : z;
            v8h d3 = inb ? src[3] : z;
            v8h* dstl = (v8h*)(&Bs[buf][bn * BS_K + bko]);
            dstl[0] = d0; dstl[1] = d1; dstl[2] = d2; dstl[3] = d3;
        } else {
            // branch-free element path (small Cin)
            _Float16 vals[32];
            #pragma unroll
            for (int u = 0; u < 32; u++) {
                int kg = k0 + bko + u;
                int kgc = kg < K - 1 ? kg : K - 1;
                int f, ci;
                if constexpr (KHW == 1) { f = 0; ci = kgc; }
                else { f = kgc / Cin; ci = kgc - f * Cin; }
                int kh = f / KW, kw = f - kh * KW;
                int ih = ihb + kh, iw = iwb + kw;
                bool inb = colOK && kg < K &&
                           ((PAD == 0) || ((unsigned)ih < (unsigned)H && (unsigned)iw < (unsigned)W));
                int ihc = clampi(ih, 0, H - 1);
                int iwc = clampi(iw, 0, W - 1);
                _Float16 v = inB[((size_t)ihc * W + iwc) * Cin + ci];
                vals[u] = inb ? v : (_Float16)0.f;
            }
            v8h* dstl = (v8h*)(&Bs[buf][bn * BS_K + bko]);
            #pragma unroll
            for (int q = 0; q < 4; q++) dstl[q] = *(v8h*)(&vals[8 * q]);
        }
    };

    auto compute = [&](int buf) {
        const _Float16* Ab = As[buf];
        const _Float16* Bb = Bs[buf];
        #pragma unroll
        for (int ks = 0; ks < KC; ks += 32) {
            const _Float16* ap  = Ab + ((mw << 4) + lm) * AS_K + ks + (lk << 4);
            const _Float16* b0p = Bb + ((nw << 5) + lm) * BS_K + ks + (lk << 4);
            const _Float16* b1p = b0p + 16 * BS_K;
            v8h alo  = *(const v8h*)(ap);
            v8h ahi  = *(const v8h*)(ap + 8);
            v8h b0lo = *(const v8h*)(b0p);
            v8h b0hi = *(const v8h*)(b0p + 8);
            v8h b1lo = *(const v8h*)(b1p);
            v8h b1hi = *(const v8h*)(b1p + 8);
            v16h a, b0, b1;
            #pragma unroll
            for (int j = 0; j < 8; j++) {
                a[j]  = alo[j];  a[j + 8]  = ahi[j];
                b0[j] = b0lo[j]; b0[j + 8] = b0hi[j];
                b1[j] = b1lo[j]; b1[j + 8] = b1hi[j];
            }
            acc0 = __builtin_amdgcn_wmma_f32_16x16x32_f16(
                false, a, false, b0, (short)0, acc0, false, false);
            acc1 = __builtin_amdgcn_wmma_f32_16x16x32_f16(
                false, a, false, b1, (short)0, acc1, false, false);
        }
    };

    stage(0, 0);
#if ASYNC_OK
    if constexpr (ANY_ASYNC) __builtin_amdgcn_s_wait_asynccnt(0);
#endif
    __syncthreads();
    for (int c = 1; c < NCH; c++) {
        stage(c & 1, c * KC);
        compute((c - 1) & 1);
#if ASYNC_OK
        if constexpr (ANY_ASYNC) __builtin_amdgcn_s_wait_asynccnt(0);
#endif
        __syncthreads();
    }
    compute((NCH - 1) & 1);

    // ---- epilogue (NHWC): contiguous co -> one v8h store per tile ----
    const int mrow = mbase + (mw << 4) + (lk << 3);
    auto store_tile = [&](const v8f& acc, int c) {
        if (c >= NTOT) return;
        if constexpr (Cout % 32 != 0) { if (mrow >= Cout) return; }
        size_t ob = (size_t)c * Cout + mrow;
        v8h rv = {};
        if (has_resid) rv = *(const v8h*)(resid + ob);
        v8h res;
        #pragma unroll
        for (int r = 0; r < 8; r++) {
            float v = acc[r] + bias[mrow + r];
            if (has_resid) v += (float)rv[r];
            if (act) v = selu_f(v);
            res[r] = (_Float16)v;
        }
        *(v8h*)(out + ob) = res;
    };
    const int c0 = nblock + (nw << 5) + lm;
    store_tile(acc0, c0);
    store_tile(acc1, c0 + 16);
}

// ---------------------------------------------------------------------------
// head kernels (tiny fraction of FLOPs); h6 is NHWC [64][49][512]
// ---------------------------------------------------------------------------
__global__ void maxpool_k(const _Float16* __restrict__ in, float* __restrict__ out,
                          int HW, int C, int NC) {
    int i = blockIdx.x * 256 + threadIdx.x;
    if (i >= NC) return;
    int n = i / C, c = i - n * C;
    const _Float16* p = in + (size_t)n * HW * C + c;
    float m = -3.4e38f;
    for (int j = 0; j < HW; j++) m = fmaxf(m, (float)p[(size_t)j * C]);
    out[i] = m;
}

__global__ void fc_k(const float* __restrict__ in, const float* __restrict__ W,
                     const float* __restrict__ b, float* __restrict__ out,
                     int N, int K, int J, int act) {
    int i = blockIdx.x * 256 + threadIdx.x;
    if (i >= N * J) return;
    int n = i / J, j = i - n * J;
    const float* row = in + (size_t)n * K;
    float s = b[j];
    for (int k = 0; k < K; k++) s += row[k] * W[(size_t)k * J + j];
    out[i] = act ? selu_f(s) : s;
}

__global__ __launch_bounds__(128)
void attn_k(const float* __restrict__ q, const float* __restrict__ k,
            const float* __restrict__ v, const float* __restrict__ f,
            float* __restrict__ g) {
    __shared__ float s[4];
    const int i = blockIdx.x;
    const int d = threadIdx.x;
    const int grp = i >> 2;

    if (d < 4) {
        int j = (grp << 2) + d;
        const float* qi = q + (size_t)i * 128;
        const float* kj = k + (size_t)j * 128;
        float dot = 0.f;
        for (int t = 0; t < 128; t++) dot += qi[t] * kj[t];
        s[d] = dot * 0.08838834764831845f;
    }
    __syncthreads();

    float m = fmaxf(fmaxf(s[0], s[1]), fmaxf(s[2], s[3]));
    float e0 = __expf(s[0] - m), e1 = __expf(s[1] - m);
    float e2 = __expf(s[2] - m), e3 = __expf(s[3] - m);
    float inv = 1.f / (e0 + e1 + e2 + e3);

    const float* vb = v + (size_t)(grp << 2) * 128 + d;
    float gv = f[(size_t)i * 128 + d] +
               (e0 * vb[0] + e1 * vb[128] + e2 * vb[256] + e3 * vb[384]) * inv;
    g[(size_t)i * 128 + d] = selu_f(gv);
}

__global__ __launch_bounds__(256)
void head_k(const float* __restrict__ g, const float* __restrict__ pw,
            const float* __restrict__ pb, float* __restrict__ out) {
    __shared__ float pooled[128];
    __shared__ float red[256];
    const int grp = blockIdx.x;
    const int j = threadIdx.x;

    if (j < 128) {
        const float* base = g + (size_t)(grp << 2) * 128 + j;
        float m = base[0];
        m = fmaxf(m, base[128]);
        m = fmaxf(m, base[256]);
        m = fmaxf(m, base[384]);
        pooled[j] = m;
    }
    __syncthreads();

    float o = pb[j];
    for (int d = 0; d < 128; d++) o += pooled[d] * pw[(size_t)d * 256 + j];
    red[j] = o * o;
    __syncthreads();
    for (int s = 128; s > 0; s >>= 1) {
        if (j < s) red[j] += red[j + s];
        __syncthreads();
    }
    float norm = sqrtf(red[0]) + 1e-9f;
    out[(size_t)grp * 256 + j] = o / norm;
}

// ---------------------------------------------------------------------------
// Host launch
// ---------------------------------------------------------------------------
enum {
    I_X = 0, I_SEG,
    I_C1W, I_C1B, I_C2W, I_C2B,
    I_R31W, I_R31B, I_R32W, I_R32B, I_R33W, I_R33B,
    I_R41W, I_R41B, I_R42W, I_R42B, I_R43W, I_R43B, I_R4SW, I_R4SB,
    I_R51W, I_R51B, I_R52W, I_R52B, I_R53W, I_R53B, I_R5SW, I_R5SB,
    I_R61W, I_R61B, I_R62W, I_R62B, I_R63W, I_R63B, I_R6SW, I_R6SB,
    I_FC2W, I_FC2B, I_FC3W, I_FC3B,
    I_WQ, I_QB, I_WK, I_KB, I_WV, I_VB,
    I_PW, I_PB
};

template <int Cin, int H, int W, int Cout, int KH, int KW, int STRIDE, int PAD>
static void conv_call(const _Float16* in, const _Float16* wt, const float* bias,
                      const _Float16* resid, _Float16* out, int act,
                      hipStream_t stream) {
    constexpr int OH = (H + 2 * PAD - KH) / STRIDE + 1;
    constexpr int OW = (W + 2 * PAD - KW) / STRIDE + 1;
    constexpr int NTOT = NIMG * OH * OW;
    dim3 grid((NTOT + 127) / 128, (Cout + 31) / 32);
    conv_wmma<Cin, H, W, Cout, KH, KW, STRIDE, PAD><<<grid, dim3(256), 0, stream>>>(
        in, wt, bias, resid ? resid : in, out, act, resid ? 1 : 0);
}

extern "C" void kernel_launch(void* const* d_in, const int* in_sizes, int n_in,
                              void* d_out, int out_size, void* d_ws, size_t ws_size,
                              hipStream_t stream) {
    (void)in_sizes; (void)n_in; (void)out_size; (void)ws_size;
    char* ws = (char*)d_ws;
    size_t off = 0;
    auto take = [&](size_t bytes) -> char* {
        char* p = ws + off;
        off += (bytes + 255) & ~(size_t)255;
        return p;
    };
    auto f16buf = [&](size_t elems) { return (_Float16*)take(elems * 2); };
    auto f32buf = [&](size_t elems) { return (float*)take(elems * 4); };

    // activation buffers (NHWC f16)
    _Float16* xh = f16buf((size_t)NIMG * 3 * 224 * 224);
    _Float16* h1 = f16buf((size_t)NIMG * 64 * 112 * 112);
    _Float16* h2 = f16buf((size_t)NIMG * 64 * 56 * 56);
    _Float16* h3 = f16buf((size_t)NIMG * 64 * 56 * 56);
    _Float16* h4 = f16buf((size_t)NIMG * 128 * 28 * 28);
    _Float16* h5 = f16buf((size_t)NIMG * 256 * 14 * 14);
    _Float16* h6 = f16buf((size_t)NIMG * 512 * 7 * 7);
    _Float16* t1 = f16buf((size_t)NIMG * 256 * 28 * 28);
    _Float16* t2 = f16buf((size_t)NIMG * 16 * 56 * 56);
    _Float16* sc = f16buf((size_t)NIMG * 128 * 28 * 28);

    // head buffers (f32)
    float* f0 = f32buf((size_t)NIMG * 512);
    float* f1 = f32buf((size_t)NIMG * 256);
    float* f2 = f32buf((size_t)NIMG * 128);
    float* qb_ = f32buf((size_t)NIMG * 128);
    float* kb_ = f32buf((size_t)NIMG * 128);
    float* vb_ = f32buf((size_t)NIMG * 128);
    float* gb_ = f32buf((size_t)NIMG * 128);

    // conv weights -> O(HW)I f16
    auto cvtw = [&](int idx, int Cout_, int Cin_, int KHW_) -> _Float16* {
        size_t n = (size_t)Cout_ * Cin_ * KHW_;
        _Float16* dst = f16buf(n);
        cvt_w_ohwi<<<(int)((n + 255) / 256), 256, 0, stream>>>(
            (const float*)d_in[idx], dst, Cin_, KHW_, (int)n);
        return dst;
    };
    _Float16* wc1 = cvtw(I_C1W, 64, 3, 25);
    _Float16* wc2 = cvtw(I_C2W, 64, 64, 25);
    _Float16* w31 = cvtw(I_R31W, 16, 64, 1);
    _Float16* w32 = cvtw(I_R32W, 16, 16, 9);
    _Float16* w33 = cvtw(I_R33W, 64, 16, 1);
    _Float16* w41 = cvtw(I_R41W, 16, 64, 1);
    _Float16* w42 = cvtw(I_R42W, 16, 16, 9);
    _Float16* w43 = cvtw(I_R43W, 128, 16, 1);
    _Float16* w4s = cvtw(I_R4SW, 128, 64, 1);
    _Float16* w51 = cvtw(I_R51W, 256, 128, 1);
    _Float16* w52 = cvtw(I_R52W, 256, 256, 9);
    _Float16* w53 = cvtw(I_R53W, 256, 256, 1);
    _Float16* w5s = cvtw(I_R5SW, 256, 128, 1);
    _Float16* w61 = cvtw(I_R61W, 512, 256, 1);
    _Float16* w62 = cvtw(I_R62W, 512, 512, 9);
    _Float16* w63 = cvtw(I_R63W, 512, 512, 1);
    _Float16* w6s = cvtw(I_R6SW, 512, 256, 1);

    {
        int n = NIMG * 3 * 224 * 224;
        cvt_x_nhwc<<<(n + 255) / 256, 256, 0, stream>>>(
            (const float*)d_in[I_X], xh, 3, 224 * 224, n);
    }

#define BIAS(idx) ((const float*)d_in[idx])

    // stem
    conv_call<3, 224, 224, 64, 5, 5, 2, 2>(xh, wc1, BIAS(I_C1B), nullptr, h1, 1, stream);
    conv_call<64, 112, 112, 64, 5, 5, 2, 2>(h1, wc2, BIAS(I_C2B), nullptr, h2, 1, stream);

    // res3 (identity shortcut)
    conv_call<64, 56, 56, 16, 1, 1, 1, 0>(h2, w31, BIAS(I_R31B), nullptr, t1, 1, stream);
    conv_call<16, 56, 56, 16, 3, 3, 1, 1>(t1, w32, BIAS(I_R32B), nullptr, t2, 1, stream);
    conv_call<16, 56, 56, 64, 1, 1, 1, 0>(t2, w33, BIAS(I_R33B), h2, h3, 1, stream);

    // res4 (projected shortcut, stride 2)
    conv_call<64, 56, 56, 16, 1, 1, 1, 0>(h3, w41, BIAS(I_R41B), nullptr, t1, 1, stream);
    conv_call<16, 56, 56, 16, 3, 3, 2, 1>(t1, w42, BIAS(I_R42B), nullptr, t2, 1, stream);
    conv_call<64, 56, 56, 128, 1, 1, 2, 0>(h3, w4s, BIAS(I_R4SB), nullptr, sc, 0, stream);
    conv_call<16, 28, 28, 128, 1, 1, 1, 0>(t2, w43, BIAS(I_R43B), sc, h4, 1, stream);

    // res5
    conv_call<128, 28, 28, 256, 1, 1, 1, 0>(h4, w51, BIAS(I_R51B), nullptr, t1, 1, stream);
    conv_call<256, 28, 28, 256, 3, 3, 2, 1>(t1, w52, BIAS(I_R52B), nullptr, t2, 1, stream);
    conv_call<128, 28, 28, 256, 1, 1, 2, 0>(h4, w5s, BIAS(I_R5SB), nullptr, sc, 0, stream);
    conv_call<256, 14, 14, 256, 1, 1, 1, 0>(t2, w53, BIAS(I_R53B), sc, h5, 1, stream);

    // res6
    conv_call<256, 14, 14, 512, 1, 1, 1, 0>(h5, w61, BIAS(I_R61B), nullptr, t1, 1, stream);
    conv_call<512, 14, 14, 512, 3, 3, 2, 1>(t1, w62, BIAS(I_R62B), nullptr, t2, 1, stream);
    conv_call<256, 14, 14, 512, 1, 1, 2, 0>(h5, w6s, BIAS(I_R6SB), nullptr, sc, 0, stream);
    conv_call<512, 7, 7, 512, 1, 1, 1, 0>(t2, w63, BIAS(I_R63B), sc, h6, 1, stream);

    // head
    maxpool_k<<<(NIMG * 512 + 255) / 256, 256, 0, stream>>>(h6, f0, 49, 512, NIMG * 512);
    fc_k<<<(NIMG * 256 + 255) / 256, 256, 0, stream>>>(
        f0, (const float*)d_in[I_FC2W], (const float*)d_in[I_FC2B], f1, NIMG, 512, 256, 1);
    fc_k<<<(NIMG * 128 + 255) / 256, 256, 0, stream>>>(
        f1, (const float*)d_in[I_FC3W], (const float*)d_in[I_FC3B], f2, NIMG, 256, 128, 1);

    fc_k<<<(NIMG * 128 + 255) / 256, 256, 0, stream>>>(
        f2, (const float*)d_in[I_WQ], (const float*)d_in[I_QB], qb_, NIMG, 128, 128, 0);
    fc_k<<<(NIMG * 128 + 255) / 256, 256, 0, stream>>>(
        f2, (const float*)d_in[I_WK], (const float*)d_in[I_KB], kb_, NIMG, 128, 128, 0);
    fc_k<<<(NIMG * 128 + 255) / 256, 256, 0, stream>>>(
        f2, (const float*)d_in[I_WV], (const float*)d_in[I_VB], vb_, NIMG, 128, 128, 0);

    attn_k<<<NIMG, 128, 0, stream>>>(qb_, kb_, vb_, f2, gb_);

    head_k<<<NGRP, 256, 0, stream>>>(
        gb_, (const float*)d_in[I_PW], (const float*)d_in[I_PB], (float*)d_out);
}
